// MultiChannelAttention_14139032338669
// MI455X (gfx1250) — matrix operand order repaired and checked by reading, not compiled, not run on hardware
//
#include <hip/hip_runtime.h>
#include <stdint.h>

typedef __bf16 bf16_t;
typedef __attribute__((ext_vector_type(16))) __bf16 v16bf;
typedef __attribute__((ext_vector_type(8)))  __bf16 v8bf;
typedef __attribute__((ext_vector_type(4)))  __bf16 v4bf;
typedef __attribute__((ext_vector_type(8)))  float  v8f;
typedef __attribute__((ext_vector_type(4)))  float  v4f;

#define B_  16
#define C_  4
#define S_  512
#define W_  512
#define H_  768

__device__ __forceinline__ v16bf combine16(v8bf lo, v8bf hi) {
  return __builtin_shufflevector(lo, hi, 0,1,2,3,4,5,6,7,8,9,10,11,12,13,14,15);
}

// Low 32 bits of a generic pointer into LDS == LDS byte offset
// (AS3->generic aperture cast keeps the LDS address in addr[31:0]).
__device__ __forceinline__ unsigned lds_off_of(const void* p) {
  return (unsigned)(uintptr_t)p;
}

// ---------------------------------------------------------------------------
// Kernel A1: hidden_state f32 -> bf16
// ---------------------------------------------------------------------------
__global__ __launch_bounds__(256)
void prep_hs_kernel(const float* __restrict__ hs, bf16_t* __restrict__ hsb, int n4) {
  int i = blockIdx.x * blockDim.x + threadIdx.x;
  if (i < n4) {
    v4f v = ((const v4f*)hs)[i];
    v4bf o;
    o[0] = (bf16_t)v[0]; o[1] = (bf16_t)v[1];
    o[2] = (bf16_t)v[2]; o[3] = (bf16_t)v[3];
    ((v4bf*)hsb)[i] = o;
  }
}

// ---------------------------------------------------------------------------
// Kernel A2: gather W_word rows (padding_idx 0 -> zeros), f32->bf16, write
// embbf[b,c,w,h] (GEMM1 B) and embT[b,c,h,w] (GEMM2 B) via LDS transpose.
// ---------------------------------------------------------------------------
__global__ __launch_bounds__(256)
void prep_emb_kernel(const int*   __restrict__ word_seq,
                     const float* __restrict__ W_word,
                     bf16_t*      __restrict__ embbf,
                     bf16_t*      __restrict__ embT)
{
  __shared__ bf16_t tile[64][66];

  const int blk = blockIdx.x;
  const int ht  = blk % 12;
  const int wt  = (blk / 12) & 7;
  const int bc  = blk / 96;
  const int w0  = wt * 64, h0 = ht * 64;
  const int t   = threadIdx.x;

  {
    const int wl   = t >> 2;
    const int hseg = (t & 3) * 16;
    const int idx  = word_seq[bc * W_ + w0 + wl];
    const float zf = (idx != 0) ? 1.0f : 0.0f;
    const v4f* src = (const v4f*)(W_word + (size_t)idx * H_ + h0 + hseg);
    v8bf o0, o1;
#pragma unroll
    for (int q = 0; q < 4; ++q) {
      v4f f = src[q];
#pragma unroll
      for (int u = 0; u < 4; ++u) {
        bf16_t v = (bf16_t)(f[u] * zf);
        tile[wl][hseg + q * 4 + u] = v;
        if (q < 2) o0[q * 4 + u] = v; else o1[(q - 2) * 4 + u] = v;
      }
    }
    bf16_t* dst = embbf + ((size_t)bc * W_ + w0 + wl) * H_ + h0 + hseg;
    *(v8bf*)(dst)     = o0;
    *(v8bf*)(dst + 8) = o1;
  }
  __syncthreads();
  {
    const int hl   = t >> 2;
    const int wseg = (t & 3) * 16;
    v8bf t0, t1;
#pragma unroll
    for (int i = 0; i < 8; ++i) t0[i] = tile[wseg + i][hl];
#pragma unroll
    for (int i = 0; i < 8; ++i) t1[i] = tile[wseg + 8 + i][hl];
    bf16_t* dst = embT + ((size_t)bc * H_ + h0 + hl) * W_ + w0 + wseg;
    *(v8bf*)(dst)     = t0;
    *(v8bf*)(dst + 8) = t1;
  }
}

// ---------------------------------------------------------------------------
// Kernel B: u = hs @ emb^T / sqrt(H);  P = exp(u)*clip(mask,0,1) (bf16),
//           denom[b,c,s] = sum_w P.
// 8 waves; wave tile = 16 rows x 64 cols (4 accumulators, shared A frag).
// Mask sub-tile is staged to LDS with global_load_async_to_lds_b128 issued
// BEFORE the K loop; s_wait_asynccnt 0 before the epilogue (per-wave region,
// no barrier needed).
// ---------------------------------------------------------------------------
__global__ __launch_bounds__(256)
void gemm1_kernel(const bf16_t* __restrict__ hsb,
                  const bf16_t* __restrict__ embbf,
                  const float*  __restrict__ mask,
                  bf16_t*       __restrict__ P,
                  float*        __restrict__ denom)
{
  __shared__ float rsum[2][64];
  __shared__ float mtile[8][16][64];   // per-wave 16x64 mask staging (32 KB)

  const int blk    = blockIdx.x;
  const int st     = blk & 7;
  const int c      = (blk >> 3) & 3;
  const int b      = blk >> 5;
  const int s0     = st * 64;
  const int lane   = threadIdx.x & 31;
  const int wv     = threadIdx.x >> 5;
  const int rowgrp = wv & 3;
  const int colgrp = wv >> 2;
  const int ln16   = lane & 15;
  const int hi     = lane >> 4;

  const float inv_temper = 0.036084391824351615f;  // 1/sqrt(768)
  const int   bc = b * C_ + c;

  const bf16_t* aBase =
      hsb + (size_t)(b * S_ + s0 + rowgrp * 16 + ln16) * H_ + hi * 16;

  float rs[8];
#pragma unroll
  for (int j = 0; j < 8; ++j) rs[j] = 0.f;

  for (int wt = 0; wt < 4; ++wt) {                 // 128 cols per iteration
    const int w0 = wt * 128 + colgrp * 64;

    // Async-stage this wave's 16x64 mask sub-tile into LDS (overlaps K loop)
#pragma unroll
    for (int ch = 0; ch < 8; ++ch) {
      const int id = lane + 32 * ch;               // 0..255 -> 16B chunks
      const int r  = id >> 4;                      // 0..15
      const int c4 = id & 15;                      // 0..15 (x 4 floats)
      const float* gsrc =
          mask + ((size_t)bc * S_ + s0 + rowgrp * 16 + r) * W_ + w0 + c4 * 4;
      unsigned ldso = lds_off_of(&mtile[wv][r][c4 * 4]);
      unsigned long long ga = (unsigned long long)(uintptr_t)gsrc;
      asm volatile("global_load_async_to_lds_b128 %0, %1, off"
                   :: "v"(ldso), "v"(ga) : "memory");
    }

    const bf16_t* bBase =
        embbf + ((size_t)bc * W_ + w0 + ln16) * H_ + hi * 16;

    v8f acc[4];
#pragma unroll
    for (int a = 0; a < 4; ++a) acc[a] = (v8f){};

#pragma unroll 2
    for (int k0 = 0; k0 < H_; k0 += 32) {
      v16bf a = combine16(*(const v8bf*)(aBase + k0),
                          *(const v8bf*)(aBase + k0 + 8));
#pragma unroll
      for (int t = 0; t < 4; ++t) {
        const bf16_t* bp = bBase + (size_t)t * 16 * H_ + k0;
        v16bf bm = combine16(*(const v8bf*)(bp), *(const v8bf*)(bp + 8));
        acc[t] = __builtin_amdgcn_wmma_f32_16x16x32_bf16(
                   false, a, false, bm, (short)0, acc[t], false, false);
      }
    }

    asm volatile("s_wait_asynccnt 0x0" ::: "memory");

    // Epilogue: D layout -> VGPR j, half hi: row = j + 8*hi, col = ln16
#pragma unroll
    for (int j = 0; j < 8; ++j) {
      const int r    = hi * 8 + j;
      const int srow = s0 + rowgrp * 16 + r;
      const size_t rowOff = ((size_t)bc * S_ + srow) * W_;
#pragma unroll
      for (int t = 0; t < 4; ++t) {
        float u = acc[t][j] * inv_temper;
        float m = mtile[wv][r][t * 16 + ln16];
        m = fminf(fmaxf(m, 0.f), 1.f);
        float p = __expf(u) * m;
        rs[j] += p;
        P[rowOff + w0 + t * 16 + ln16] = (bf16_t)p;
      }
    }
  }

  // Row sums: reduce over the 16 lanes of each half, combine colgroups.
#pragma unroll
  for (int j = 0; j < 8; ++j) {
    float v = rs[j];
    v += __shfl_xor(v, 1, 32);
    v += __shfl_xor(v, 2, 32);
    v += __shfl_xor(v, 4, 32);
    v += __shfl_xor(v, 8, 32);
    rs[j] = v;
  }
  if (ln16 == 0) {
#pragma unroll
    for (int j = 0; j < 8; ++j)
      rsum[colgrp][rowgrp * 16 + hi * 8 + j] = rs[j];
  }
  __syncthreads();
  if (threadIdx.x < 64)
    denom[(size_t)bc * S_ + s0 + threadIdx.x] =
        rsum[0][threadIdx.x] + rsum[1][threadIdx.x];
}

// ---------------------------------------------------------------------------
// Kernel C: out[b,s,c*H+h] = cw[b,c]/(denom+1e-10) * sum_w P[s,w]*emb[w,h]
// Wave tile = 16 rows x 64 h-cols (4 accumulators). H loop: 6 x 128 cols.
// ---------------------------------------------------------------------------
__global__ __launch_bounds__(256)
void gemm2_kernel(const int*    __restrict__ channel_ids,
                  const float*  __restrict__ W_channel,
                  const bf16_t* __restrict__ P,
                  const bf16_t* __restrict__ embT,
                  const float*  __restrict__ denom,
                  float*        __restrict__ out)
{
  const int blk    = blockIdx.x;
  const int st     = blk & 7;
  const int c      = (blk >> 3) & 3;
  const int b      = blk >> 5;
  const int s0     = st * 64;
  const int lane   = threadIdx.x & 31;
  const int wv     = threadIdx.x >> 5;
  const int rowgrp = wv & 3;
  const int colgrp = wv >> 2;
  const int ln16   = lane & 15;
  const int hi     = lane >> 4;
  const int bc     = b * C_ + c;

  // Channel softmax over C=4
  float wc[4];
  float mx = -1e30f;
#pragma unroll
  for (int cc = 0; cc < 4; ++cc) {
    wc[cc] = W_channel[channel_ids[b * C_ + cc]];
    mx = fmaxf(mx, wc[cc]);
  }
  float se = 0.f;
#pragma unroll
  for (int cc = 0; cc < 4; ++cc) { wc[cc] = __expf(wc[cc] - mx); se += wc[cc]; }
  const float cw = wc[c] / se;

  float dinv[8];
#pragma unroll
  for (int j = 0; j < 8; ++j) {
    const int srow = s0 + rowgrp * 16 + hi * 8 + j;
    dinv[j] = cw / (denom[(size_t)bc * S_ + srow] + 1e-10f);
  }

  const bf16_t* aBase =
      P + ((size_t)bc * S_ + s0 + rowgrp * 16 + ln16) * W_ + hi * 16;

  for (int hc = 0; hc < 6; ++hc) {                 // 128 h-cols per iteration
    const int h0 = hc * 128 + colgrp * 64;
    const bf16_t* bBase =
        embT + ((size_t)bc * H_ + h0 + ln16) * W_ + hi * 16;

    v8f acc[4];
#pragma unroll
    for (int a = 0; a < 4; ++a) acc[a] = (v8f){};

#pragma unroll 2
    for (int k0 = 0; k0 < W_; k0 += 32) {          // K = W dimension
      v16bf a = combine16(*(const v8bf*)(aBase + k0),
                          *(const v8bf*)(aBase + k0 + 8));
#pragma unroll
      for (int t = 0; t < 4; ++t) {
        const bf16_t* bp = bBase + (size_t)t * 16 * W_ + k0;
        v16bf bm = combine16(*(const v8bf*)(bp), *(const v8bf*)(bp + 8));
        acc[t] = __builtin_amdgcn_wmma_f32_16x16x32_bf16(
                   false, a, false, bm, (short)0, acc[t], false, false);
      }
    }

#pragma unroll
    for (int j = 0; j < 8; ++j) {
      const int srow = s0 + rowgrp * 16 + hi * 8 + j;
      float* orow = out + ((size_t)b * S_ + srow) * (C_ * H_) + c * H_;
#pragma unroll
      for (int t = 0; t < 4; ++t)
        orow[h0 + t * 16 + ln16] = acc[t][j] * dinv[j];
    }
  }
}

// ---------------------------------------------------------------------------
extern "C" void kernel_launch(void* const* d_in, const int* in_sizes, int n_in,
                              void* d_out, int out_size, void* d_ws, size_t ws_size,
                              hipStream_t stream) {
  const int*   word_seq  = (const int*)  d_in[0];
  const float* hidden    = (const float*)d_in[1];
  const float* mask      = (const float*)d_in[2];
  const int*   chan_ids  = (const int*)  d_in[3];
  const float* W_word    = (const float*)d_in[4];
  const float* W_channel = (const float*)d_in[5];
  float* out = (float*)d_out;

  // Workspace layout (bytes):
  //   hsb   @ 0          : B*S*H   bf16 = 12,582,912
  //   P     @ 12,582,912 : B*C*S*W bf16 = 33,554,432
  //   denom @ 46,137,344 : B*C*S   f32  =    131,072
  //   embbf @ 46,268,416 : B*C*W*H bf16 = 50,331,648
  //   embT  @ 96,600,064 : B*C*H*W bf16 = 50,331,648
  char* ws = (char*)d_ws;
  bf16_t* hsb   = (bf16_t*)(ws);
  bf16_t* P     = (bf16_t*)(ws + 12582912);
  float*  denom = (float*) (ws + 46137344);
  bf16_t* embbf = (bf16_t*)(ws + 46268416);
  bf16_t* embT  = (bf16_t*)(ws + 96600064);

  const int n4 = (B_ * S_ * H_) / 4;
  prep_hs_kernel <<<(n4 + 255) / 256, 256, 0, stream>>>(hidden, hsb, n4);
  prep_emb_kernel<<<B_ * C_ * 8 * 12, 256, 0, stream>>>(word_seq, W_word,
                                                        embbf, embT);
  gemm1_kernel<<<B_ * C_ * (S_ / 64), 256, 0, stream>>>(hsb, embbf, mask,
                                                        P, denom);
  gemm2_kernel<<<B_ * C_ * (S_ / 64), 256, 0, stream>>>(chan_ids, W_channel,
                                                        P, embT, denom, out);
  (void)in_sizes; (void)n_in; (void)out_size; (void)ws_size;
}
